// LGeMSelfAttention_28887950033771
// MI455X (gfx1250) — matrix operand order repaired and checked by reading, not compiled, not run on hardware
//
#include <hip/hip_runtime.h>

// ---------------- problem constants ----------------
#define Bn 2
#define Tn 2048
#define Cn 2048
#define Hn 16
#define Dn 128
#define BTn (Bn * Tn)   // 4096

// ---------------- WMMA types ----------------
typedef __attribute__((ext_vector_type(16))) __bf16 v16bf;
typedef __attribute__((ext_vector_type(8)))  float  v8f;
typedef __attribute__((ext_vector_type(4)))  unsigned u32x4;
typedef __attribute__((ext_vector_type(8)))  unsigned u32x8;

union FragB { unsigned u[8]; v16bf v; };
union FragC { float    f[8]; v8f   v; };

static __device__ inline v8f wmma_bf16(const FragB& a, const FragB& b, v8f c) {
  // D = A(16x32 bf16) * B(32x16 bf16) + C(16x16 f32)
  return __builtin_amdgcn_wmma_f32_16x16x32_bf16(false, a.v, false, b.v,
                                                 (short)0, c, false, false);
}

// A-matrix (16x32, 16-bit) VGPR i / lane layout (ISA 7.12.2):
static __device__ inline int a_koff(int i, int lane) {
  return 2 * (i & 3) + ((i >= 4) ? 16 : 0) + ((lane >= 16) ? 8 : 0);
}
// B-matrix (32x16, 16-bit): lane n = lane&15 ; packed pair starts at:
static __device__ inline int b_koff(int i, int lane) {
  return 2 * (i & 3) + ((i >= 4) ? 16 : 0) + ((lane >= 16) ? 8 : 0);
}

// f32 -> bf16 round-to-nearest-even
static __device__ inline unsigned short f2bf(float f) {
  unsigned x = __float_as_uint(f);
  unsigned r = x + 0x7FFFu + ((x >> 16) & 1u);
  return (unsigned short)(r >> 16);
}

// ---------------- Tensor Data Mover (TDM) helpers ----------------
// Low 32 bits of a generic (aperture-based) LDS address = byte offset in LDS.
static __device__ inline unsigned lds_off(const void* p) {
  return (unsigned)(unsigned long long)p;
}

// 2D tile load, 2-byte elements, optional LDS row padding.
// tile_d0/tensor_d0/stride_d0 along the contiguous dimension (elements).
static __device__ inline void tdm_load_2d(unsigned lds_addr, const void* gptr,
                                          unsigned tile_d0, unsigned tile_d1,
                                          unsigned tensor_d0,
                                          unsigned tensor_d1,
                                          unsigned long long stride_d0,
                                          unsigned pad_interval_code,
                                          unsigned pad_amount_code,
                                          unsigned pad_en) {
  const unsigned long long ga = (unsigned long long)gptr;
  u32x4 g0;
  g0[0] = 1u;                                   // count=1, user descriptor
  g0[1] = lds_addr;                             // D#.lds_addr
  g0[2] = (unsigned)(ga & 0xFFFFFFFFull);       // global_addr[31:0]
  g0[3] = ((unsigned)(ga >> 32) & 0x01FFFFFFu)  // global_addr[56:32]
          | (2u << 30);                         // type = 2 (image)

  // group1 bits (ISA 8.4), assembled as 4x u64
  const unsigned long long td0 = tensor_d0, td1 = tensor_d1;
  const unsigned long long q0 =
      (1ull << 16)                                    // data_size = 2 bytes
      | ((unsigned long long)(pad_en & 1u) << 20)     // pad_enable
      | ((unsigned long long)(pad_interval_code & 7u) << 22)
      | ((unsigned long long)(pad_amount_code & 0x7Fu) << 25)
      | ((td0 & 0xFFFFull) << 48);                    // tensor_dim0[15:0]
  const unsigned long long q1 =
      ((td0 >> 16) & 0xFFFFull)                       // tensor_dim0[31:16]
      | ((td1 & 0xFFFFFFFFull) << 16)                 // tensor_dim1
      | ((unsigned long long)(tile_d0 & 0xFFFFu) << 48);  // tile_dim0
  const unsigned long long q2 =
      (unsigned long long)(tile_d1 & 0xFFFFu)         // tile_dim1
      | ((stride_d0 & 0xFFFFFFFFull) << 32);          // dim0_stride[31:0]
  const unsigned long long q3 =
      ((stride_d0 >> 32) & 0xFFFFull);                // dim0_stride[47:32]
  u32x8 g1;
  g1[0] = (unsigned)q0; g1[1] = (unsigned)(q0 >> 32);
  g1[2] = (unsigned)q1; g1[3] = (unsigned)(q1 >> 32);
  g1[4] = (unsigned)q2; g1[5] = (unsigned)(q2 >> 32);
  g1[6] = (unsigned)q3; g1[7] = (unsigned)(q3 >> 32);
  u32x4 gz; gz[0] = 0; gz[1] = 0; gz[2] = 0; gz[3] = 0;  // 2D: groups 2/3 unused
  asm volatile("tensor_load_to_lds %0, %1, %2, %3"
               :
               : "s"(g0), "s"(g1), "s"(gz), "s"(gz)
               : "memory");
}

// =====================================================================
// Kernel 1: QKV projections (bf16 WMMA GEMM) + RoPE factor epilogue.
// grid = (N/128, M/128, 3) ; block = 256 (8 waves, 16 rows x 128 cols each)
// z=0 -> Q [b,h,t,d], z=1 -> K [b,h,t,d], z=2 -> V transposed [b,h,d,t]
// =====================================================================
__global__ __launch_bounds__(256) void k_qkv(
    const float* __restrict__ X,                     // [BT, C] f32
    const float* __restrict__ Wq, const float* __restrict__ Wk,
    const float* __restrict__ Wv,
    unsigned short* __restrict__ qws, unsigned short* __restrict__ kws,
    unsigned short* __restrict__ vws) {
  const int z = blockIdx.z;
  const float* W = (z == 0) ? Wq : (z == 1) ? Wk : Wv;
  const int n0 = blockIdx.x * 128;
  const int m0 = blockIdx.y * 128;
  const int tid = threadIdx.x;
  const int lane = tid & 31;
  const int w = tid >> 5;

  __shared__ unsigned short As[128][34];   // A tile [m][k] bf16, padded
  __shared__ unsigned short Bt[128][34];   // W tile transposed [n][k], padded

  FragC c[8];
  for (int f = 0; f < 8; ++f)
    for (int r = 0; r < 8; ++r) c[f].f[r] = 0.0f;

  for (int k0 = 0; k0 < Cn; k0 += 32) {
    for (int idx = tid; idx < 128 * 32; idx += 256) {
      int m = idx >> 5, kk = idx & 31;
      As[m][kk] = f2bf(X[(size_t)(m0 + m) * Cn + k0 + kk]);
    }
    for (int idx = tid; idx < 32 * 128; idx += 256) {
      int kk = idx >> 7, n = idx & 127;
      Bt[n][kk] = f2bf(W[(size_t)(k0 + kk) * Cn + n0 + n]);
    }
    __syncthreads();

    FragB a;
    const int m = (w << 4) + (lane & 15);
    for (int i = 0; i < 8; ++i)
      a.u[i] = *(const unsigned*)&As[m][a_koff(i, lane)];
    for (int nb = 0; nb < 8; ++nb) {
      FragB b;
      const int n = (nb << 4) + (lane & 15);
      for (int i = 0; i < 8; ++i)
        b.u[i] = *(const unsigned*)&Bt[n][b_koff(i, lane)];
      c[nb].v = wmma_bf16(a, b, c[nb].v);
    }
    __syncthreads();
  }

  const int rbase = m0 + (w << 4) + ((lane >> 4) << 3);
  for (int nb = 0; nb < 8; ++nb) {
    const int ng = n0 + (nb << 4) + (lane & 15);
    const int h = ng >> 7;
    const int dd = ng & 127;
    const int fi = dd & 63;
    const float freq = expf(-(float)(2 * fi) * (9.210340371976184f / 128.0f));
    for (int r = 0; r < 8; ++r) {
      const int mg = rbase + r;
      const int bb = mg >> 11;
      const int t = mg & 2047;
      float val = c[nb].f[r];
      if (z < 2) {
        float s, cth;
        sincosf((float)t * freq, &s, &cth);
        val *= (cth + s);
      }
      const unsigned short bv = f2bf(val);
      if (z == 0)
        qws[(((size_t)bb * Hn + h) * Tn + t) * Dn + dd] = bv;
      else if (z == 1)
        kws[(((size_t)bb * Hn + h) * Tn + t) * Dn + dd] = bv;
      else
        vws[(((size_t)bb * Hn + h) * Dn + dd) * Tn + t] = bv;
    }
  }
}

// =====================================================================
// Kernel 2: flash attention; TDM double-buffered K/V staging + WMMA.
// grid = (T/64, B*H) ; block = 128 (4 waves, each wave = 16 q rows)
// =====================================================================
__global__ __launch_bounds__(128) void k_attn(
    const unsigned short* __restrict__ qws,   // [b,h,t,d] bf16
    const unsigned short* __restrict__ kws,   // [b,h,t,d] bf16
    const unsigned short* __restrict__ vws,   // [b,h,d,t] bf16
    const float* __restrict__ mask,           // [b,1,t,t] f32
    unsigned short* __restrict__ aows) {      // [b,t,h*d] bf16
  const int bh = blockIdx.y;
  const int b = bh >> 4;
  const int h = bh & 15;
  const int tid = threadIdx.x, lane = tid & 31, w = tid >> 5;
  const int qt = blockIdx.x * 64 + (w << 4);

  const unsigned short* qp = qws + (size_t)bh * Tn * Dn;
  const unsigned short* kp = kws + (size_t)bh * Tn * Dn;
  const unsigned short* vp = vws + (size_t)bh * Dn * Tn;
  const float* mp = mask + (size_t)b * Tn * Tn;

  // double-buffered TDM destinations; pad rows by 1 dword via D#.pad
  __shared__ unsigned short kbuf[2][32][130];  // K block [key][d]
  __shared__ unsigned short vbuf[2][128][34];  // V^T block [d][t]
  __shared__ float          sbuf[4][16][33];   // per-wave S scratch
  __shared__ unsigned short pbuf[4][16][34];   // per-wave P (bf16)
  __shared__ float          abuf[4][16];       // per-wave row broadcast

  FragC o[8];
  for (int f = 0; f < 8; ++f)
    for (int r = 0; r < 8; ++r) o[f].f[r] = 0.0f;
  float mstate = -3.0e38f, lstate = 0.0f;

  // Q fragments for this wave's 16 rows (K = 128 in 4 steps of 32)
  FragB aq[4];
  {
    const unsigned short* qrow = qp + (size_t)(qt + (lane & 15)) * Dn;
    for (int kb = 0; kb < 4; ++kb)
      for (int i = 0; i < 8; ++i)
        aq[kb].u[i] = *(const unsigned*)(qrow + kb * 32 + a_koff(i, lane));
  }

  const float scale = 0.02209708691207961f;  // 1/sqrt(2048)
  const int NJ = Tn / 32;

  // prologue: wave 0 DMAs block 0 into buffer 0
  if (w == 0) {
    // K tile: 128 contiguous elems/row, 32 rows; LDS pad 1 dw per 64 dw
    tdm_load_2d(lds_off(&kbuf[0][0][0]), kp, 128, 32, Dn, Tn, Dn, 5, 0, 1);
    // V^T tile: 32 contiguous elems/row, 128 rows; LDS pad 1 dw per 16 dw
    tdm_load_2d(lds_off(&vbuf[0][0][0]), vp, 32, 128, Tn, Dn, Tn, 3, 0, 1);
  }

  for (int j = 0; j < NJ; ++j) {
    const int buf = j & 1;
    if (w == 0) {
      if (j + 1 < NJ) {
        const int nxt = buf ^ 1;
        tdm_load_2d(lds_off(&kbuf[nxt][0][0]),
                    kp + (size_t)(j + 1) * 32 * Dn, 128, 32, Dn, Tn, Dn, 5, 0,
                    1);
        tdm_load_2d(lds_off(&vbuf[nxt][0][0]), vp + (size_t)(j + 1) * 32, 32,
                    128, Tn, Dn, Tn, 3, 0, 1);
        __builtin_amdgcn_s_wait_tensorcnt(2);  // block j's pair complete
      } else {
        __builtin_amdgcn_s_wait_tensorcnt(0);
      }
    }
    __syncthreads();  // publish buffer `buf` to all waves

    // ---- S = Q * K^T (two 16x16 halves), scale + mask, to LDS ----
    for (int half = 0; half < 2; ++half) {
      FragC sfr;
      for (int r = 0; r < 8; ++r) sfr.f[r] = 0.0f;
      const int key = (half << 4) + (lane & 15);
      for (int kb = 0; kb < 4; ++kb) {
        FragB bk;
        for (int i = 0; i < 8; ++i)
          bk.u[i] =
              *(const unsigned*)&kbuf[buf][key][kb * 32 + b_koff(i, lane)];
        sfr.v = wmma_bf16(aq[kb], bk, sfr.v);
      }
      const int col = j * 32 + (half << 4) + (lane & 15);
      const int rsub = ((lane >> 4) << 3);
      for (int r = 0; r < 8; ++r) {
        const float sv =
            sfr.f[r] * scale + mp[(size_t)(qt + rsub + r) * Tn + col];
        sbuf[w][rsub + r][(half << 4) + (lane & 15)] = sv;
      }
    }
    __syncthreads();

    // ---- online softmax: lanes 0..15 own one row each ----
    if (lane < 16) {
      float rm = mstate;
      for (int c2 = 0; c2 < 32; ++c2) rm = fmaxf(rm, sbuf[w][lane][c2]);
      const float alpha = __expf(mstate - rm);
      float ls = 0.0f;
      for (int c2 = 0; c2 < 32; ++c2) {
        const float p = __expf(sbuf[w][lane][c2] - rm);
        ls += p;
        pbuf[w][lane][c2] = f2bf(p);
      }
      lstate = lstate * alpha + ls;
      mstate = rm;
      abuf[w][lane] = alpha;
    }
    __syncthreads();

    // ---- rescale O, then O += P * V via WMMA ----
    const int rsub = ((lane >> 4) << 3);
    for (int r = 0; r < 8; ++r) {
      const float av = abuf[w][rsub + r];
      for (int f = 0; f < 8; ++f) o[f].f[r] *= av;
    }
    FragB ap;
    for (int i = 0; i < 8; ++i)
      ap.u[i] = *(const unsigned*)&pbuf[w][lane & 15][a_koff(i, lane)];
    for (int f = 0; f < 8; ++f) {
      FragB bv;
      for (int i = 0; i < 8; ++i)
        bv.u[i] = *(const unsigned*)&vbuf[buf][(f << 4) + (lane & 15)]
                                         [b_koff(i, lane)];
      o[f].v = wmma_bf16(ap, bv, o[f].v);
    }
    __syncthreads();  // all readers done before buffer is re-filled
  }

  // ---- finalize: divide by row sums, store bf16 [b,t,h*d] ----
  if (lane < 16) abuf[w][lane] = 1.0f / lstate;
  __syncthreads();
  const int rsub = ((lane >> 4) << 3);
  for (int r = 0; r < 8; ++r) {
    const int trow = qt + rsub + r;
    const float inv = abuf[w][rsub + r];
    for (int f = 0; f < 8; ++f) {
      const int dd = (f << 4) + (lane & 15);
      aows[(size_t)(b * Tn + trow) * Cn + h * Dn + dd] = f2bf(o[f].f[r] * inv);
    }
  }
}

// =====================================================================
// Kernel 3: output projection (bf16 A x bf16 W -> f32 out)
// grid = (N/128, M/128) ; block = 256
// =====================================================================
__global__ __launch_bounds__(256) void k_oproj(
    const unsigned short* __restrict__ A,   // [BT, C] bf16
    const float* __restrict__ W,            // [C, C] f32
    float* __restrict__ out) {              // [BT, C] f32
  const int n0 = blockIdx.x * 128;
  const int m0 = blockIdx.y * 128;
  const int tid = threadIdx.x;
  const int lane = tid & 31;
  const int w = tid >> 5;

  __shared__ unsigned short As[128][34];
  __shared__ unsigned short Bt[128][34];

  FragC c[8];
  for (int f = 0; f < 8; ++f)
    for (int r = 0; r < 8; ++r) c[f].f[r] = 0.0f;

  for (int k0 = 0; k0 < Cn; k0 += 32) {
    for (int idx = tid; idx < 128 * 16; idx += 256) {
      const int m = idx >> 4, c2 = idx & 15;
      *(unsigned*)&As[m][c2 * 2] =
          ((const unsigned*)(A + (size_t)(m0 + m) * Cn + k0))[c2];
    }
    for (int idx = tid; idx < 32 * 128; idx += 256) {
      const int kk = idx >> 7, n = idx & 127;
      Bt[n][kk] = f2bf(W[(size_t)(k0 + kk) * Cn + n0 + n]);
    }
    __syncthreads();

    FragB a;
    const int m = (w << 4) + (lane & 15);
    for (int i = 0; i < 8; ++i)
      a.u[i] = *(const unsigned*)&As[m][a_koff(i, lane)];
    for (int nb = 0; nb < 8; ++nb) {
      FragB b;
      const int n = (nb << 4) + (lane & 15);
      for (int i = 0; i < 8; ++i)
        b.u[i] = *(const unsigned*)&Bt[n][b_koff(i, lane)];
      c[nb].v = wmma_bf16(a, b, c[nb].v);
    }
    __syncthreads();
  }

  const int rbase = m0 + (w << 4) + ((lane >> 4) << 3);
  for (int nb = 0; nb < 8; ++nb) {
    const int ng = n0 + (nb << 4) + (lane & 15);
    for (int r = 0; r < 8; ++r)
      out[(size_t)(rbase + r) * Cn + ng] = c[nb].f[r];
  }
}

// =====================================================================
extern "C" void kernel_launch(void* const* d_in, const int* in_sizes, int n_in,
                              void* d_out, int out_size, void* d_ws,
                              size_t ws_size, hipStream_t stream) {
  (void)in_sizes; (void)n_in; (void)out_size; (void)ws_size;
  const float* X    = (const float*)d_in[0];  // [2,2048,2048]
  const float* mask = (const float*)d_in[1];  // [2,1,2048,2048]
  const float* Wq   = (const float*)d_in[2];
  const float* Wk   = (const float*)d_in[3];
  const float* Wv   = (const float*)d_in[4];
  const float* Wo   = (const float*)d_in[5];

  const size_t seg = (size_t)BTn * Cn * sizeof(unsigned short);  // 16 MB
  char* ws = (char*)d_ws;
  unsigned short* qws  = (unsigned short*)(ws + 0 * seg);
  unsigned short* kws  = (unsigned short*)(ws + 1 * seg);
  unsigned short* vws  = (unsigned short*)(ws + 2 * seg);
  unsigned short* aows = (unsigned short*)(ws + 3 * seg);

  k_qkv<<<dim3(Cn / 128, BTn / 128, 3), 256, 0, stream>>>(X, Wq, Wk, Wv, qws,
                                                          kws, vws);
  k_attn<<<dim3(Tn / 64, Bn * Hn), 128, 0, stream>>>(qws, kws, vws, mask,
                                                     aows);
  k_oproj<<<dim3(Cn / 128, BTn / 128), 256, 0, stream>>>(aows, Wo,
                                                         (float*)d_out);
}
